// TernBinLayer_50225347559963
// MI455X (gfx1250) — 8-bit matrix operand order repaired and checked by reading, compiled, not run on hardware
//
#include <hip/hip_runtime.h>


typedef int v8i __attribute__((ext_vector_type(8)));

#define NROWS 131072
#define KDIM  512
#define NDIM  512
#define MTILES (NROWS / 16)
#define KTILES (KDIM / 64)
#define NTILES (NDIM / 16)

__global__ __launch_bounds__(256) void pack_x_kernel(const float* __restrict__ x,
                                                     int* __restrict__ xp) {
    unsigned idx   = blockIdx.x * 256u + threadIdx.x;
    unsigned v     = idx & 7u;
    unsigned lane  = (idx >> 3) & 31u;
    unsigned ktile = (idx >> 8) & 7u;
    unsigned mtile = idx >> 11;

    unsigned kbase = ((v >> 1) << 4) | ((v & 1u) << 2);
    unsigned row   = mtile * 16u + (lane & 15u);
    unsigned kcol  = ktile * 64u + kbase + ((lane >> 4) << 3);

    const float4 f = *(const float4*)(x + (size_t)row * KDIM + kcol);
    unsigned b0 = (f.x >= 0.0f) ? 0x01u : 0xFFu;
    unsigned b1 = (f.y >= 0.0f) ? 0x01u : 0xFFu;
    unsigned b2 = (f.z >= 0.0f) ? 0x01u : 0xFFu;
    unsigned b3 = (f.w >= 0.0f) ? 0x01u : 0xFFu;
    xp[idx] = (int)(b0 | (b1 << 8) | (b2 << 16) | (b3 << 24));
}

__global__ __launch_bounds__(256) void pack_w_kernel(const float* __restrict__ W,
                                                     int* __restrict__ wp) {
    unsigned idx   = blockIdx.x * 256u + threadIdx.x;
    unsigned v     = idx & 7u;
    unsigned lane  = (idx >> 3) & 31u;
    unsigned ktile = (idx >> 8) & 7u;
    unsigned ntile = idx >> 11;

    unsigned kbase = ((v >> 1) << 4) | ((v & 1u) << 2) | ((lane >> 4) << 3);
    unsigned n     = ntile * 16u + (lane & 15u);
    unsigned k0    = ktile * 64u + kbase;

    unsigned packed = 0u;
#pragma unroll
    for (int b = 0; b < 4; ++b) {
        float f = W[(size_t)(k0 + b) * NDIM + n];
        int   q = (f > 0.5f) ? 1 : ((f < -0.5f) ? -1 : 0);
        packed |= ((unsigned)(q & 0xFF)) << (8 * b);
    }
    wp[idx] = (int)packed;
}

__global__ __launch_bounds__(256) void tern_gemm_kernel(const int* __restrict__ xp,
                                                        const int* __restrict__ wp,
                                                        const int* __restrict__ bshift,
                                                        float* __restrict__ out) {
    const int shift = *bshift;

    const unsigned lane   = threadIdx.x & 31u;
    const unsigned widx   = threadIdx.x >> 5;
    const unsigned gwave  = blockIdx.x * 8u + widx;
    const unsigned ngroup = gwave & 7u;
    const unsigned mtile  = gwave >> 3;

    v8i acc0 = {0, 0, 0, 0, 0, 0, 0, 0};
    v8i acc1 = {0, 0, 0, 0, 0, 0, 0, 0};
    v8i acc2 = {0, 0, 0, 0, 0, 0, 0, 0};
    v8i acc3 = {0, 0, 0, 0, 0, 0, 0, 0};

    const int* aBase = xp + (size_t)mtile * (KTILES * 256u) + lane * 8u;
    const int* wBase = wp + lane * 8u;
    const unsigned nt0 = ngroup * 4u;

#pragma unroll
    for (int kt = 0; kt < KTILES; ++kt) {
        const v8i a = *(const v8i*)(aBase + kt * 256);
        if (kt + 1 < KTILES) {
            __builtin_prefetch(aBase + (kt + 1) * 256, 0, 1);
        }
        const v8i b0 = *(const v8i*)(wBase + (size_t)(nt0 + 0) * (KTILES * 256u) + kt * 256);
        const v8i b1 = *(const v8i*)(wBase + (size_t)(nt0 + 1) * (KTILES * 256u) + kt * 256);
        const v8i b2 = *(const v8i*)(wBase + (size_t)(nt0 + 2) * (KTILES * 256u) + kt * 256);
        const v8i b3 = *(const v8i*)(wBase + (size_t)(nt0 + 3) * (KTILES * 256u) + kt * 256);

        acc0 = __builtin_amdgcn_wmma_i32_16x16x64_iu8(true, a, true, b0, acc0, false, false);
        acc1 = __builtin_amdgcn_wmma_i32_16x16x64_iu8(true, a, true, b1, acc1, false, false);
        acc2 = __builtin_amdgcn_wmma_i32_16x16x64_iu8(true, a, true, b2, acc2, false, false);
        acc3 = __builtin_amdgcn_wmma_i32_16x16x64_iu8(true, a, true, b3, acc3, false, false);
    }

    const unsigned rowBase = mtile * 16u + ((lane >> 4) << 3);
    const unsigned colBase = ngroup * 64u + (lane & 15u);
    float* o = out + (size_t)rowBase * NDIM + colBase;

#pragma unroll
    for (int r = 0; r < 8; ++r) {
        float* po = o + (size_t)r * NDIM;
        po[0]  = ((acc0[r] >> shift) >= 0) ? 1.0f : -1.0f;
        po[16] = ((acc1[r] >> shift) >= 0) ? 1.0f : -1.0f;
        po[32] = ((acc2[r] >> shift) >= 0) ? 1.0f : -1.0f;
        po[48] = ((acc3[r] >> shift) >= 0) ? 1.0f : -1.0f;
    }
}

extern "C" void kernel_launch(void* const* d_in, const int* in_sizes, int n_in,
                              void* d_out, int out_size, void* d_ws, size_t ws_size,
                              hipStream_t stream) {
    const float* x  = (const float*)d_in[0];
    const float* W  = (const float*)d_in[1];
    const int* bsh  = (const int*)d_in[2];
    float* out      = (float*)d_out;

    int* xp = (int*)d_ws;
    int* wp = xp + (size_t)MTILES * KTILES * 256;

    pack_x_kernel<<<(MTILES * KTILES * 256) / 256, 256, 0, stream>>>(x, xp);
    pack_w_kernel<<<(NTILES * KTILES * 256) / 256, 256, 0, stream>>>(W, wp);
    tern_gemm_kernel<<<(MTILES * (NTILES / 4)) / 8, 256, 0, stream>>>(xp, wp, bsh, out);
}
